// BiLSTM_CRF_52639119179919
// MI455X (gfx1250) — matrix operand order repaired and checked by reading, not compiled, not run on hardware
//
#include <hip/hip_runtime.h>
#include <cstdint>
#include <cstddef>

// ---------------- problem constants (match reference) ----------------
#define TT 512      // sequence length
#define BB 32       // batch
#define EE 256      // embedding
#define HH 256      // hidden
#define G4 1024     // 4*H gate width
#define KK 22       // tags + start + stop
#define START_TAG 20
#define STOP_TAG  21
#define MROWS (TT * BB)   // 16384 rows for the big GEMMs

typedef __attribute__((ext_vector_type(16))) __bf16 v16bf;
typedef __attribute__((ext_vector_type(8)))  float  v8f;

__device__ __forceinline__ float sigf(float x) { return 1.0f / (1.0f + __expf(-x)); }

// ============ CDNA5 wave32 WMMA fragment loaders (CDNA5 ISA §7.12.2) ============

// A-matrix 16x32 bf16 fragment from a row-major source (row stride ld elements).
// lanes 0-15: M=lane,   K = {0..7, 16..23} ; lanes 16-31: M=lane-16, K = {8..15, 24..31}
__device__ __forceinline__ v16bf load_a16(const __bf16* src, int row0, int k0, int ld, int lane) {
  const int g = lane >> 4;
  const int m = lane & 15;
  const __bf16* p = src + (size_t)(row0 + m) * ld + k0 + g * 8;
  v16bf a;
#pragma unroll
  for (int e = 0; e < 8; ++e) a[e] = p[e];           // K = g*8 + e
#pragma unroll
  for (int e = 0; e < 8; ++e) a[e + 8] = p[16 + e];  // K = 16 + g*8 + e
  return a;
}

// B-matrix 32x16 bf16 fragment where B[k][n] = W[n][k], W row-major (row stride ld).
__device__ __forceinline__ v16bf load_b16(const __bf16* W, int n0, int k0, int ld, int lane) {
  const int n = lane & 15;
  const int g = lane >> 4;
  const __bf16* p = W + (size_t)(n0 + n) * ld + k0 + g * 8;
  v16bf b;
#pragma unroll
  for (int e = 0; e < 8; ++e) b[e] = p[e];           // K = g*8 + e
#pragma unroll
  for (int e = 0; e < 8; ++e) b[e + 8] = p[16 + e];  // K = 16 + g*8 + e
  return b;
}

// Async Global -> LDS 16-byte copy (gfx1250 GLOBAL_LOAD_ASYNC_TO_LDS_B128, ASYNCcnt).
__device__ __forceinline__ void async_g2l_b128(unsigned lds_byte_addr, const void* gaddr) {
  asm volatile("global_load_async_to_lds_b128 %0, %1, off"
               :: "v"(lds_byte_addr), "v"(gaddr) : "memory");
}
__device__ __forceinline__ void wait_asynccnt0() {
  asm volatile("s_wait_asynccnt 0x0" ::: "memory");
}

// ============================== kernels ==============================

// fp32 -> bf16 weight conversion (done once per launch; weights are tiny)
__global__ void cvt_f32_to_bf16(const float* __restrict__ src, __bf16* __restrict__ dst, int n) {
  int i = blockIdx.x * blockDim.x + threadIdx.x;
  if (i < n) dst[i] = (__bf16)src[i];
}

// x[t][b][:] = emb[word[b][t]][:]  (transposed gather, cast to bf16)
__global__ void gather_embed(const int* __restrict__ word, const float* __restrict__ emb,
                             __bf16* __restrict__ x_bf) {
  int id = blockIdx.x * 256 + threadIdx.x;   // TT*BB*EE total
  int e  = id & (EE - 1);
  int rb = id >> 8;                          // t*B + b
  int b  = rb & (BB - 1);
  int t  = rb >> 5;
  int w  = word[b * TT + t];
  x_bf[(size_t)rb * EE + e] = (__bf16)emb[(size_t)w * EE + e];
}

// out[M][N] = X[M][Kd] @ W[N][Kd]^T + b1[N] + b2[N]   (bf16 in, f32 out, WMMA)
// 256 threads = 8 waves; one 32x32 output supertile (2x2 WMMA tiles) per wave.
__global__ void wmma_gemm_bias(const __bf16* __restrict__ X, const __bf16* __restrict__ W,
                               const float* __restrict__ b1, const float* __restrict__ b2,
                               float* __restrict__ out, int M, int N, int Kd) {
  const int lane = threadIdx.x & 31;
  const int wave = threadIdx.x >> 5;
  const int tiles_n2 = N >> 5;               // 32-wide supertiles
  const int gt   = blockIdx.x * 8 + wave;
  const int row0 = (gt / tiles_n2) * 32;
  const int col0 = (gt % tiles_n2) * 32;
  const int g  = lane >> 4;
  const int ln = lane & 15;

  v8f acc[2][2];
#pragma unroll
  for (int ni = 0; ni < 2; ++ni) {
    const int n = col0 + ni * 16 + ln;
    const float bias = b1[n] + b2[n];
#pragma unroll
    for (int mi = 0; mi < 2; ++mi)
#pragma unroll
      for (int e = 0; e < 8; ++e) acc[mi][ni][e] = bias;
  }

  const int ksteps = Kd >> 5;
  for (int ks = 0; ks < ksteps; ++ks) {
    // near-temporal prefetch of the next A k-slab (global_prefetch_b8)
    __builtin_prefetch(X + (size_t)(row0 + ln) * Kd + (size_t)(ks + 1) * 32, 0, 3);
    v16bf a0 = load_a16(X, row0,      ks * 32, Kd, lane);
    v16bf a1 = load_a16(X, row0 + 16, ks * 32, Kd, lane);
    v16bf b0 = load_b16(W, col0,      ks * 32, Kd, lane);
    v16bf b1f = load_b16(W, col0 + 16, ks * 32, Kd, lane);
    acc[0][0] = __builtin_amdgcn_wmma_f32_16x16x32_bf16(false, a0, false, b0, (short)0,
                                                        acc[0][0], false, false);
    acc[1][0] = __builtin_amdgcn_wmma_f32_16x16x32_bf16(false, a1, false, b0, (short)0,
                                                        acc[1][0], false, false);
    acc[0][1] = __builtin_amdgcn_wmma_f32_16x16x32_bf16(false, a0, false, b1f, (short)0,
                                                        acc[0][1], false, false);
    acc[1][1] = __builtin_amdgcn_wmma_f32_16x16x32_bf16(false, a1, false, b1f, (short)0,
                                                        acc[1][1], false, false);
  }
#pragma unroll
  for (int mi = 0; mi < 2; ++mi)
#pragma unroll
    for (int ni = 0; ni < 2; ++ni)
#pragma unroll
      for (int e = 0; e < 8; ++e) {
        const int m = row0 + mi * 16 + g * 8 + e;
        out[(size_t)m * N + col0 + ni * 16 + ln] = acc[mi][ni][e];
      }
}

// Persistent bidirectional LSTM layer: gridDim.x == 2 (block 0 = fwd, 1 = rev),
// 1024 threads = 32 waves. Per step: g = xW[s] + h @ whh^T (WMMA), then gates.
// h (bf16) / c (f32) / gates live in LDS; the NEXT step's 128 KB xW slab is
// staged into one reused LDS buffer with GLOBAL_LOAD_ASYNC_TO_LDS_B128 overlapped with
// the elementwise phase, keeping the sequential critical path off global latency.
__global__ void lstm_bidir(const float* __restrict__ xw_f, const float* __restrict__ xw_r,
                           const __bf16* __restrict__ whh_f, const __bf16* __restrict__ whh_r,
                           __bf16* __restrict__ out_bf, float* __restrict__ out_f32) {
  extern __shared__ char smem[];
  __bf16* h_bf  = (__bf16*)smem;                                   // 16 KB
  float*  c_st  = (float*)(smem + BB * HH * 2);                    // 32 KB
  float*  gates = (float*)(smem + BB * HH * 2 + BB * HH * 4);      // 128 KB
  float*  xw_st = (float*)(smem + BB * HH * 2 + BB * HH * 4 + BB * G4 * 4); // 128 KB stage

  const int dir = blockIdx.x;
  const float*  xw  = dir ? xw_r : xw_f;
  const __bf16* whh = dir ? whh_r : whh_f;
  const int col_off = dir * HH;

  const int tid  = threadIdx.x;
  const int wave = tid >> 5;      // owns gate columns [32*wave, 32*wave+32)
  const int lane = tid & 31;
  const int g    = lane >> 4;
  const int ln   = lane & 15;

  for (int i = tid; i < BB * HH; i += 1024) { h_bf[i] = (__bf16)0.0f; c_st[i] = 0.0f; }

  // pre-stage step 0's xW slab (8 x b128 per thread = 128 KB total)
  {
    const int s0 = dir ? (TT - 1) : 0;
    const float* src = xw + (size_t)s0 * BB * G4;
#pragma unroll
    for (int r = 0; r < 8; ++r) {
      const int idx = tid + r * 1024;                 // 16-byte chunk index
      async_g2l_b128((unsigned)(size_t)(xw_st + idx * 4), src + (size_t)idx * 4);
    }
  }
  wait_asynccnt0();
  __syncthreads();

  for (int t = 0; t < TT; ++t) {
    const int s = dir ? (TT - 1 - t) : t;

    // accumulators: 2 M-tiles x 2 N-tiles, seeded with staged xW (LDS)
    v8f acc[2][2];
#pragma unroll
    for (int mt = 0; mt < 2; ++mt)
#pragma unroll
      for (int nt = 0; nt < 2; ++nt) {
        const int n = (2 * wave + nt) * 16 + ln;
#pragma unroll
        for (int e = 0; e < 8; ++e) {
          const int m = mt * 16 + g * 8 + e;
          acc[mt][nt][e] = xw_st[m * G4 + n];
        }
      }

    // h @ whh^T : K = 256 in 8 bf16 WMMA k-steps; whh slabs are L2-resident
    for (int ks = 0; ks < 8; ++ks) {
      v16bf a0 = load_a16(h_bf, 0,  ks * 32, HH, lane);
      v16bf a1 = load_a16(h_bf, 16, ks * 32, HH, lane);
      v16bf w0 = load_b16(whh, (2 * wave + 0) * 16, ks * 32, HH, lane);
      v16bf w1 = load_b16(whh, (2 * wave + 1) * 16, ks * 32, HH, lane);
      acc[0][0] = __builtin_amdgcn_wmma_f32_16x16x32_bf16(false, a0, false, w0, (short)0,
                                                          acc[0][0], false, false);
      acc[1][0] = __builtin_amdgcn_wmma_f32_16x16x32_bf16(false, a1, false, w0, (short)0,
                                                          acc[1][0], false, false);
      acc[0][1] = __builtin_amdgcn_wmma_f32_16x16x32_bf16(false, a0, false, w1, (short)0,
                                                          acc[0][1], false, false);
      acc[1][1] = __builtin_amdgcn_wmma_f32_16x16x32_bf16(false, a1, false, w1, (short)0,
                                                          acc[1][1], false, false);
    }

    // stage gate pre-activations to LDS
#pragma unroll
    for (int mt = 0; mt < 2; ++mt)
#pragma unroll
      for (int nt = 0; nt < 2; ++nt) {
        const int n = (2 * wave + nt) * 16 + ln;
#pragma unroll
        for (int e = 0; e < 8; ++e) {
          const int m = mt * 16 + g * 8 + e;
          gates[m * G4 + n] = acc[mt][nt][e];
        }
      }
    __syncthreads();   // acc-init reads of xw_st are complete; gates visible

    // kick off async staging of the NEXT step's xW slab (overlaps elementwise)
    if (t + 1 < TT) {
      const int sn = dir ? (TT - 2 - t) : (t + 1);
      const float* src = xw + (size_t)sn * BB * G4;
#pragma unroll
      for (int r = 0; r < 8; ++r) {
        const int idx = tid + r * 1024;
        async_g2l_b128((unsigned)(size_t)(xw_st + idx * 4), src + (size_t)idx * 4);
      }
    }

    // elementwise LSTM cell update: 32x256 = 8192 states over 1024 threads
#pragma unroll
    for (int r = 0; r < 8; ++r) {
      const int idx = tid + r * 1024;
      const int b   = idx >> 8;
      const int hu  = idx & (HH - 1);
      const float gi = gates[b * G4 + hu];
      const float gf = gates[b * G4 + HH + hu];
      const float gc = gates[b * G4 + 2 * HH + hu];
      const float go = gates[b * G4 + 3 * HH + hu];
      const float cn = sigf(gf) * c_st[idx] + sigf(gi) * tanhf(gc);
      const float hn = sigf(go) * tanhf(cn);
      c_st[idx] = cn;
      h_bf[idx] = (__bf16)hn;
      const size_t o = ((size_t)s * BB + b) * (2 * HH) + col_off + hu;
      out_bf[o]  = (__bf16)hn;
      out_f32[o] = hn;
    }
    wait_asynccnt0();
    __syncthreads();   // h/c updated AND next xW slab landed
  }
}

// feats[r][k] = l1[r][:] @ w_out[:,k] + b_out[k]   (K=22 too skinny for WMMA)
__global__ void out_proj(const float* __restrict__ l1, const float* __restrict__ w_out,
                         const float* __restrict__ b_out, float* __restrict__ feats) {
  const int row = blockIdx.x * 8 + (threadIdx.x >> 5);
  const int k   = threadIdx.x & 31;
  if (k >= KK) return;
  const float* x = l1 + (size_t)row * (2 * HH);
  float s = b_out[k];
  for (int j = 0; j < 2 * HH; ++j) s += x[j] * w_out[j * KK + k];
  feats[(size_t)row * KK + k] = s;
}

// CRF forward scan + gold score + reduction. Single workgroup; alpha is only 32x22.
__global__ void crf_score(const float* __restrict__ feats, const float* __restrict__ trans,
                          const int* __restrict__ mask, const int* __restrict__ tags,
                          float* __restrict__ out) {
  __shared__ float tr[KK * KK];
  __shared__ float alpha[BB * KK];
  __shared__ float res[BB];
  const int tid = threadIdx.x;
  for (int i = tid; i < KK * KK; i += 1024) tr[i] = trans[i];
  const bool act = tid < BB * KK;
  const int b = tid / KK;
  const int k = tid % KK;
  __syncthreads();
  if (act) alpha[tid] = feats[b * KK + k] + tr[START_TAG * KK + k];
  __syncthreads();
  for (int t = 1; t < TT; ++t) {
    float nv = 0.0f;
    if (act) {
      float mx = -3.0e38f;
#pragma unroll
      for (int j = 0; j < KK; ++j) mx = fmaxf(mx, alpha[b * KK + j] + tr[j * KK + k]);
      float sm = 0.0f;
#pragma unroll
      for (int j = 0; j < KK; ++j) sm += __expf(alpha[b * KK + j] + tr[j * KK + k] - mx);
      nv = mx + __logf(sm) + feats[((size_t)t * BB + b) * KK + k];
      if (mask[b * TT + t] == 0) nv = alpha[b * KK + k];
    }
    __syncthreads();
    if (act) alpha[b * KK + k] = nv;
    __syncthreads();
  }
  if (tid < BB) {
    float mx = -3.0e38f;
    for (int j = 0; j < KK; ++j) mx = fmaxf(mx, alpha[tid * KK + j] + tr[j * KK + STOP_TAG]);
    float sm = 0.0f;
    for (int j = 0; j < KK; ++j) sm += __expf(alpha[tid * KK + j] + tr[j * KK + STOP_TAG] - mx);
    const float logZ = mx + __logf(sm);

    float emit = 0.0f, trs = 0.0f;
    int len = 0, prev = 0;
    for (int t = 0; t < TT; ++t) {
      const int m  = mask[tid * TT + t];
      const int tg = tags[tid * TT + t];
      if (t == 0) trs = tr[START_TAG * KK + tg];
      else if (m) trs += tr[prev * KK + tg];
      if (m) { emit += feats[((size_t)t * BB + tid) * KK + tg]; ++len; }
      prev = tg;
    }
    const int last = tags[tid * TT + (len > 0 ? len - 1 : 0)];
    res[tid] = logZ - (emit + trs + tr[last * KK + STOP_TAG]);
  }
  __syncthreads();
  if (tid == 0) {
    float s = 0.0f;
    for (int bb = 0; bb < BB; ++bb) s += res[bb];
    out[0] = s;
  }
}

// ============================== launcher ==============================

extern "C" void kernel_launch(void* const* d_in, const int* in_sizes, int n_in,
                              void* d_out, int out_size, void* d_ws, size_t ws_size,
                              hipStream_t stream) {
  (void)in_sizes; (void)n_in; (void)out_size; (void)ws_size;

  const int*   word  = (const int*)  d_in[0];
  const int*   mask  = (const int*)  d_in[1];
  const int*   tags  = (const int*)  d_in[2];
  const float* emb   = (const float*)d_in[3];
  const float* wih0f = (const float*)d_in[4];
  const float* whh0f = (const float*)d_in[5];
  const float* bih0f = (const float*)d_in[6];
  const float* bhh0f = (const float*)d_in[7];
  const float* wih0r = (const float*)d_in[8];
  const float* whh0r = (const float*)d_in[9];
  const float* bih0r = (const float*)d_in[10];
  const float* bhh0r = (const float*)d_in[11];
  const float* wih1f = (const float*)d_in[12];
  const float* whh1f = (const float*)d_in[13];
  const float* bih1f = (const float*)d_in[14];
  const float* bhh1f = (const float*)d_in[15];
  const float* wih1r = (const float*)d_in[16];
  const float* whh1r = (const float*)d_in[17];
  const float* bih1r = (const float*)d_in[18];
  const float* bhh1r = (const float*)d_in[19];
  const float* w_out = (const float*)d_in[20];
  const float* b_out = (const float*)d_in[21];
  const float* trans = (const float*)d_in[22];

  // ---- workspace carve-up ----
  char* ws = (char*)d_ws;
  auto carve = [&](size_t bytes) -> char* {
    char* p = ws;
    ws += (bytes + 255) & ~(size_t)255;
    return p;
  };
  __bf16* x_bf     = (__bf16*)carve((size_t)MROWS * EE * 2);     // 8 MB
  __bf16* wih0f_b  = (__bf16*)carve((size_t)G4 * EE * 2);
  __bf16* whh0f_b  = (__bf16*)carve((size_t)G4 * HH * 2);
  __bf16* wih0r_b  = (__bf16*)carve((size_t)G4 * EE * 2);
  __bf16* whh0r_b  = (__bf16*)carve((size_t)G4 * HH * 2);
  __bf16* wih1f_b  = (__bf16*)carve((size_t)G4 * 2 * HH * 2);
  __bf16* whh1f_b  = (__bf16*)carve((size_t)G4 * HH * 2);
  __bf16* wih1r_b  = (__bf16*)carve((size_t)G4 * 2 * HH * 2);
  __bf16* whh1r_b  = (__bf16*)carve((size_t)G4 * HH * 2);
  float*  xw_f     = (float*)carve((size_t)MROWS * G4 * 4);      // 64 MB
  float*  xw_r     = (float*)carve((size_t)MROWS * G4 * 4);      // 64 MB
  __bf16* l_bf     = (__bf16*)carve((size_t)MROWS * 2 * HH * 2); // 16 MB
  float*  l_f32    = (float*)carve((size_t)MROWS * 2 * HH * 4);  // 32 MB
  float*  feats    = (float*)carve((size_t)MROWS * KK * 4);      // 1.4 MB

  // ---- 1. weight conversion to bf16 ----
  auto cvt = [&](const float* s, __bf16* d, int n) {
    cvt_f32_to_bf16<<<(n + 255) / 256, 256, 0, stream>>>(s, d, n);
  };
  cvt(wih0f, wih0f_b, G4 * EE);
  cvt(whh0f, whh0f_b, G4 * HH);
  cvt(wih0r, wih0r_b, G4 * EE);
  cvt(whh0r, whh0r_b, G4 * HH);
  cvt(wih1f, wih1f_b, G4 * 2 * HH);
  cvt(whh1f, whh1f_b, G4 * HH);
  cvt(wih1r, wih1r_b, G4 * 2 * HH);
  cvt(whh1r, whh1r_b, G4 * HH);

  // ---- 2. embedding gather (transposed to [T][B][E], bf16) ----
  gather_embed<<<(MROWS * EE) / 256, 256, 0, stream>>>(word, emb, x_bf);

  const int gemm_blocks = (MROWS / 32) * (G4 / 32) / 8;  // 2048 (32x32 supertiles)
  const size_t lstm_smem = (size_t)BB * HH * 2 + (size_t)BB * HH * 4
                         + (size_t)BB * G4 * 4 + (size_t)BB * G4 * 4;  // 304 KB

  // ---- 3. layer 0: input GEMMs (WMMA) + bidirectional recurrence ----
  wmma_gemm_bias<<<gemm_blocks, 256, 0, stream>>>(x_bf, wih0f_b, bih0f, bhh0f, xw_f,
                                                  MROWS, G4, EE);
  wmma_gemm_bias<<<gemm_blocks, 256, 0, stream>>>(x_bf, wih0r_b, bih0r, bhh0r, xw_r,
                                                  MROWS, G4, EE);
  lstm_bidir<<<2, 1024, lstm_smem, stream>>>(xw_f, xw_r, whh0f_b, whh0r_b, l_bf, l_f32);

  // ---- 4. layer 1 (input = concat fwd/rev of layer 0, Kd = 512) ----
  wmma_gemm_bias<<<gemm_blocks, 256, 0, stream>>>(l_bf, wih1f_b, bih1f, bhh1f, xw_f,
                                                  MROWS, G4, 2 * HH);
  wmma_gemm_bias<<<gemm_blocks, 256, 0, stream>>>(l_bf, wih1r_b, bih1r, bhh1r, xw_r,
                                                  MROWS, G4, 2 * HH);
  lstm_bidir<<<2, 1024, lstm_smem, stream>>>(xw_f, xw_r, whh1f_b, whh1r_b, l_bf, l_f32);

  // ---- 5. emission features + CRF ----
  out_proj<<<MROWS / 8, 256, 0, stream>>>(l_f32, w_out, b_out, feats);
  crf_score<<<1, 1024, 0, stream>>>(feats, trans, mask, tags, (float*)d_out);
}
